// Qwen3MoEDecoderLayer_8581344658119
// MI455X (gfx1250) — matrix operand order repaired and checked by reading, not compiled, not run on hardware
//
#include <hip/hip_runtime.h>
#include <hip/hip_bf16.h>

// ---------------------------------------------------------------------------
// Qwen3-MoE decoder layer for MI455X (gfx1250, wave32, WMMA f32_16x16x32_f16,
// async global->LDS tile copies)
// ---------------------------------------------------------------------------

typedef __attribute__((ext_vector_type(16))) _Float16 v16h;
typedef __attribute__((ext_vector_type(8)))  float    v8f;

constexpr int T_TOK = 4096;   // B*S tokens
constexpr int S_    = 1024;
constexpr int H_    = 2048;
constexpr int NH_   = 32;
constexpr int NKV_  = 4;
constexpr int D_    = 128;
constexpr int E_    = 32;
constexpr int KTOP  = 8;
constexpr int I_    = 768;
constexpr int IS_   = 4096;
constexpr int B_    = 4;

union AFrag { v16h v; uint32_t u[8]; };

__device__ inline v8f vz8() {
  v8f z;
#pragma unroll
  for (int i = 0; i < 8; ++i) z[i] = 0.f;
  return z;
}

__device__ inline v8f wmma32(v16h a, v16h b, v8f c) {
  return __builtin_amdgcn_wmma_f32_16x16x32_f16(false, a, false, b,
                                                (short)0, c, false, false);
}

// Async DMA: global -> LDS, 16 bytes per lane, tracked with ASYNCcnt.
__device__ inline void async_g2l_b128(uint32_t ldsaddr, const void* gaddr) {
  asm volatile("global_load_async_to_lds_b128 %0, %1, off"
               :: "v"(ldsaddr), "v"(gaddr) : "memory");
}
__device__ inline void wait_asynccnt0() {
  asm volatile("s_wait_asynccnt 0x0" ::: "memory");
}

// A fragment (16x32 f16, ISA 7.12.2): lane m=lane&15, kh=lane>>4;
// vgpr j<4 : k = 8*kh + 2j ; j>=4 : k = 16 + 8*kh + 2(j-4)
__device__ inline v16h ld_afrag_k(const _Float16* s, int ld, int k0) {
  int lane = threadIdx.x & 31;
  int m = lane & 15, kh = lane >> 4;
  const uint32_t* p = (const uint32_t*)(s + (long)m * ld) + (k0 >> 1);
  AFrag f;
#pragma unroll
  for (int j = 0; j < 4; ++j) {
    f.u[j]     = p[4 * kh + j];
    f.u[4 + j] = p[8 + 4 * kh + j];
  }
  return f.v;
}

// B fragment (32x16 f16): lane n=lane&15, kh=lane>>4;
// vgpr j<4 : k = 8*kh + 2j ; j>=4 : k = 16 + 8*kh + 2(j-4)
// source tile stored [n][k] (k contiguous)
__device__ inline v16h ld_bfrag_k(const _Float16* s, int ld, int k0) {
  int lane = threadIdx.x & 31;
  int n = lane & 15, kh = lane >> 4;
  const uint32_t* p = (const uint32_t*)(s + (long)n * ld) + (k0 >> 1);
  AFrag f;
#pragma unroll
  for (int j = 0; j < 4; ++j) {
    f.u[j]     = p[4 * kh + j];
    f.u[4 + j] = p[8 + 4 * kh + j];
  }
  return f.v;
}

// ---------------------------------------------------------------------------
// Generic WMMA GEMM:  C[M,N](f32) = A[M,K](f16) * B[K,N](f32 -> f16 in LDS)
// A tile is copied with global_load_async_to_lds_b128 (no VGPR round trip);
// B tile goes through VALU for the f32->f16 conversion.
// AG: 0 = plain rows; 1 = expert token-gather rows; 2 = expert compact rows
// EPI: 0 = store f32; 1 = resid + acc -> f32; 2 = store f16;
//      3 = acc * row_weight scatter-atomicAdd into Cf[token]
// ---------------------------------------------------------------------------
template<int AG, int EPI>
__global__ __launch_bounds__(256) void gemm_kernel(
    const _Float16* __restrict__ A, int lda,
    const float* __restrict__ Bw, long strideB,
    float* __restrict__ Cf, _Float16* __restrict__ Ch, int ldc,
    const float* __restrict__ resid,
    const int* __restrict__ gidx, const float* __restrict__ gw,
    const int* __restrict__ offs,
    int M, int N, int K)
{
  __shared__ _Float16 As[128 * 48];   // [m][k] k-contiguous, 96B row (16B align)
  __shared__ _Float16 Bs[128 * 40];   // [n][k] k-contiguous, pad 8
  int e = (AG ? blockIdx.z : 0);
  int rbase = 0, Mrows = M;
  if (AG) {
    rbase = offs[e];
    Mrows = offs[e + 1] - rbase;
    if ((int)(blockIdx.x * 128) >= Mrows) return;
  }
  const float* B = Bw + (long)e * strideB;
  int m0 = blockIdx.x * 128, n0 = blockIdx.y * 128;
  int tid = threadIdx.x, w = tid >> 5, lane = tid & 31;
  int wm = w & 3, wn = w >> 2;           // 4x2 waves -> 128x128 tile
  int nL = lane & 15, kh = lane >> 4;
  uint32_t asbase = (uint32_t)(uintptr_t)(void*)&As[0];

  v8f acc[2][4];
#pragma unroll
  for (int mi = 0; mi < 2; ++mi)
#pragma unroll
    for (int ni = 0; ni < 4; ++ni) acc[mi][ni] = vz8();

  for (int k0 = 0; k0 < K; k0 += 32) {
    // A tile: 128 rows x 32B = 256 b128 async DMA transfers (1 per thread)
    {
      int m = tid >> 1, seg = tid & 1;
      int gm = m0 + m;
      uint32_t lds = asbase + (uint32_t)(m * 96 + seg * 16);
      if (gm < Mrows) {
        long row = (AG == 1) ? (long)gidx[rbase + gm]
                 : (AG == 2) ? (long)(rbase + gm) : (long)gm;
        async_g2l_b128(lds, A + row * (long)lda + k0 + seg * 8);
      } else {
#pragma unroll
        for (int j = 0; j < 4; ++j)
          *(uint32_t*)(As + m * 48 + seg * 8 + 2 * j) = 0u;
      }
    }
    // B tile: 32x128 f32 -> f16, transpose to [n][k]
    for (int i = tid; i < 32 * 128; i += 256) {
      int kk = i >> 7, n = i & 127;
      float bv = 0.f;
      if (n0 + n < N) bv = B[(long)(k0 + kk) * N + (n0 + n)];
      Bs[n * 40 + kk] = (_Float16)bv;
    }
    wait_asynccnt0();
    __syncthreads();
    if (k0 + 32 < K)
      __builtin_prefetch(B + (long)(k0 + 32) * N + n0 + (tid & 127), 0, 1);

    v16h bf[4];
#pragma unroll
    for (int ni = 0; ni < 4; ++ni)
      bf[ni] = ld_bfrag_k(Bs + (64 * wn + 16 * ni) * 40, 40, 0);
#pragma unroll
    for (int mi = 0; mi < 2; ++mi) {
      v16h af = ld_afrag_k(As + (32 * wm + 16 * mi) * 48, 48, 0);
#pragma unroll
      for (int ni = 0; ni < 4; ++ni)
        acc[mi][ni] = wmma32(af, bf[ni], acc[mi][ni]);
    }
    __syncthreads();
  }

#pragma unroll
  for (int mi = 0; mi < 2; ++mi)
#pragma unroll
    for (int ni = 0; ni < 4; ++ni)
#pragma unroll
      for (int r = 0; r < 8; ++r) {
        int ml  = 32 * wm + 16 * mi + r + 8 * kh;
        int gmr = m0 + ml;
        int col = n0 + 64 * wn + 16 * ni + nL;
        if (gmr < Mrows && col < N) {
          float v = acc[mi][ni][r];
          if (EPI == 0) {
            long orow = AG ? (long)(rbase + gmr) : (long)gmr;
            Cf[orow * (long)ldc + col] = v;
          } else if (EPI == 1) {
            Cf[(long)gmr * ldc + col] = resid[(long)gmr * ldc + col] + v;
          } else if (EPI == 2) {
            long orow = AG ? (long)(rbase + gmr) : (long)gmr;
            Ch[orow * (long)ldc + col] = (_Float16)v;
          } else {
            int   tok = gidx[rbase + gmr];
            float wt  = gw[rbase + gmr];
            unsafeAtomicAdd(&Cf[(long)tok * ldc + col], v * wt);
          }
        }
      }
}

// ---------------------------------------------------------------------------
// Flash attention with WMMA for QK^T and PV. 128 q rows / block, 64-key tiles.
// K tile streamed with async global->LDS DMA; V tile transposed via VALU.
// q16/k16/v16 layouts: [T, nH, D] f16. Output [T, NH*D] f16.
// ---------------------------------------------------------------------------
__global__ __launch_bounds__(256) void attn_kernel(
    const _Float16* __restrict__ Q, const _Float16* __restrict__ Kg,
    const _Float16* __restrict__ Vg, _Float16* __restrict__ O)
{
  __shared__ _Float16 Kt[64][136];     // [key][d], 272B row (16B align)
  __shared__ _Float16 Vt[128][72];     // [d][key] (transposed)
  __shared__ _Float16 Pw[8][16][72];   // per-wave P staging [m][key]
  int qt = blockIdx.x, h = blockIdx.y, b = blockIdx.z;
  int kvh = h >> 3;                    // NH/NKV = 8
  int q0 = qt * 128;
  int tid = threadIdx.x, w = tid >> 5, lane = tid & 31;
  int nL = lane & 15, kh = lane >> 4;
  const float scale = 0.088388347648318447f;  // D^-0.5
  uint32_t ktbase = (uint32_t)(uintptr_t)(void*)&Kt[0][0];

  v16h qf[4];
  {
    const _Float16* qb = Q + ((long)(b * S_ + q0 + 16 * w) * NH_ + h) * D_;
#pragma unroll
    for (int kk = 0; kk < 4; ++kk) qf[kk] = ld_afrag_k(qb, NH_ * D_, kk * 32);
  }
  v8f of[8];
#pragma unroll
  for (int dd = 0; dd < 8; ++dd) of[dd] = vz8();
  float mrun[8], lrun[8];
#pragma unroll
  for (int r = 0; r < 8; ++r) { mrun[r] = -1e30f; lrun[r] = 0.f; }

  int ntiles = (q0 + 128) / 64;
  for (int t = 0; t < ntiles; ++t) {
    int kv0 = t * 64;
    // K tile: 64 keys x 256B = 1024 b128 async DMA transfers (4 per thread)
    for (int i = tid; i < 64 * 16; i += 256) {
      int key = i >> 4, seg = i & 15;
      long tokk = (long)b * S_ + kv0 + key;
      async_g2l_b128(ktbase + (uint32_t)(key * 272 + seg * 16),
                     Kg + ((long)tokk * NKV_ + kvh) * D_ + seg * 8);
    }
    for (int i = tid; i < 64 * 64; i += 256) {      // V tile transposed
      int key = i >> 6, dp = i & 63;
      uint32_t vv = ((const uint32_t*)(Vg +
          ((long)(b * S_ + kv0 + key) * NKV_ + kvh) * D_))[dp];
      union { uint32_t u; _Float16 hh[2]; } cv; cv.u = vv;
      Vt[2 * dp][key]     = cv.hh[0];
      Vt[2 * dp + 1][key] = cv.hh[1];
    }
    wait_asynccnt0();
    __syncthreads();

    v8f sf[4];
#pragma unroll
    for (int ni = 0; ni < 4; ++ni) sf[ni] = vz8();
#pragma unroll
    for (int kk = 0; kk < 4; ++kk)
#pragma unroll
      for (int ni = 0; ni < 4; ++ni) {
        v16h bf = ld_bfrag_k(&Kt[ni * 16][0], 136, kk * 32);
        sf[ni] = wmma32(qf[kk], bf, sf[ni]);
      }

    float rowmax[8];
#pragma unroll
    for (int r = 0; r < 8; ++r) rowmax[r] = -1e30f;
#pragma unroll
    for (int ni = 0; ni < 4; ++ni)
#pragma unroll
      for (int r = 0; r < 8; ++r) {
        int keyg = kv0 + ni * 16 + nL;
        int qg   = q0 + 16 * w + r + 8 * kh;
        float s  = sf[ni][r] * scale;
        sf[ni][r] = s;
        if (keyg <= qg && s > rowmax[r]) rowmax[r] = s;
      }
#pragma unroll
    for (int mm = 1; mm < 16; mm <<= 1)
#pragma unroll
      for (int r = 0; r < 8; ++r)
        rowmax[r] = fmaxf(rowmax[r], __shfl_xor(rowmax[r], mm, 32));

    float alpha[8], rowsum[8];
#pragma unroll
    for (int r = 0; r < 8; ++r) {
      float mnew = fmaxf(mrun[r], rowmax[r]);
      alpha[r] = __expf(mrun[r] - mnew);
      mrun[r] = mnew;
      rowsum[r] = 0.f;
    }
#pragma unroll
    for (int ni = 0; ni < 4; ++ni)
#pragma unroll
      for (int r = 0; r < 8; ++r) {
        int keyg = kv0 + ni * 16 + nL;
        int qg   = q0 + 16 * w + r + 8 * kh;
        float p = (keyg <= qg) ? __expf(sf[ni][r] - mrun[r]) : 0.f;
        rowsum[r] += p;
        Pw[w][r + 8 * kh][ni * 16 + nL] = (_Float16)p;
      }
#pragma unroll
    for (int mm = 1; mm < 16; mm <<= 1)
#pragma unroll
      for (int r = 0; r < 8; ++r)
        rowsum[r] += __shfl_xor(rowsum[r], mm, 32);
#pragma unroll
    for (int r = 0; r < 8; ++r) lrun[r] = lrun[r] * alpha[r] + rowsum[r];
#pragma unroll
    for (int dd = 0; dd < 8; ++dd)
#pragma unroll
      for (int r = 0; r < 8; ++r) of[dd][r] *= alpha[r];

#pragma unroll
    for (int kk2 = 0; kk2 < 2; ++kk2) {
      v16h pa = ld_afrag_k(&Pw[w][0][0], 72, kk2 * 32);
#pragma unroll
      for (int dd = 0; dd < 8; ++dd) {
        v16h bv = ld_bfrag_k(&Vt[dd * 16][0], 72, kk2 * 32);
        of[dd] = wmma32(pa, bv, of[dd]);
      }
    }
    __syncthreads();
  }

#pragma unroll
  for (int dd = 0; dd < 8; ++dd)
#pragma unroll
    for (int r = 0; r < 8; ++r) {
      int m = r + 8 * kh;
      long tok = (long)b * S_ + q0 + 16 * w + m;
      float val = of[dd][r] / lrun[r];
      O[tok * (long)(NH_ * D_) + (long)h * D_ + dd * 16 + nL] = (_Float16)val;
    }
}

// ---------------------------------------------------------------------------
// Elementwise / reduction kernels
// ---------------------------------------------------------------------------
__global__ __launch_bounds__(256) void rmsnorm_kernel(
    const float* __restrict__ x, const float* __restrict__ wv,
    _Float16* __restrict__ y, int Hn)
{
  __shared__ float red[8];
  long base = (long)blockIdx.x * Hn;
  float ss = 0.f;
  for (int i = threadIdx.x; i < Hn; i += 256) { float v = x[base + i]; ss += v * v; }
  for (int mm = 1; mm < 32; mm <<= 1) ss += __shfl_xor(ss, mm, 32);
  if ((threadIdx.x & 31) == 0) red[threadIdx.x >> 5] = ss;
  __syncthreads();
  float tot = 0.f;
  for (int i = 0; i < 8; ++i) tot += red[i];
  float rn = rsqrtf(tot / (float)Hn + 1e-6f);
  for (int i = threadIdx.x; i < Hn; i += 256)
    y[base + i] = (_Float16)(x[base + i] * rn * wv[i]);
}

__global__ __launch_bounds__(128) void qk_rope_kernel(
    const float* __restrict__ q32, const float* __restrict__ k32,
    const float* __restrict__ qnw, const float* __restrict__ knw,
    _Float16* __restrict__ q16, _Float16* __restrict__ k16)
{
  __shared__ float xs[128];
  __shared__ float red[4];
  int s = blockIdx.x, hh = blockIdx.y, b = blockIdx.z;
  bool isq = hh < NH_;
  int h  = isq ? hh : hh - NH_;
  int nh = isq ? NH_ : NKV_;
  long tok = (long)b * S_ + s;
  const float* src = (isq ? q32 : k32) + ((long)tok * nh + h) * D_;
  const float* wv  = isq ? qnw : knw;
  _Float16* dst    = (isq ? q16 : k16) + ((long)tok * nh + h) * D_;
  int d = threadIdx.x;
  float x = src[d];
  float ss = x * x;
  for (int mm = 1; mm < 32; mm <<= 1) ss += __shfl_xor(ss, mm, 32);
  if ((d & 31) == 0) red[d >> 5] = ss;
  __syncthreads();
  float tot = red[0] + red[1] + red[2] + red[3];
  float rn = rsqrtf(tot / 128.f + 1e-6f);
  float xn = x * rn * wv[d];
  xs[d] = xn;
  __syncthreads();
  int i2 = d & 63;
  float ang = (float)s * __powf(1.0e6f, -(float)(2 * i2) / 128.f);
  float sn, cs;
  __sincosf(ang, &sn, &cs);
  float rot = (d < 64) ? -xs[d + 64] : xs[d - 64];
  dst[d] = (_Float16)(xn * cs + rot * sn);
}

__global__ void cvt16_kernel(const float* __restrict__ x,
                             _Float16* __restrict__ y, long n)
{
  long i = (long)blockIdx.x * blockDim.x + threadIdx.x;
  if (i < n) y[i] = (_Float16)x[i];
}

__global__ void router_kernel(const float* __restrict__ logits,
                              float* __restrict__ topw, int* __restrict__ topi,
                              int* __restrict__ counts)
{
  int t = blockIdx.x * blockDim.x + threadIdx.x;
  if (t >= T_TOK) return;
  float l[E_];
  float mx = -1e30f;
  for (int e = 0; e < E_; ++e) { l[e] = logits[(long)t * E_ + e]; mx = fmaxf(mx, l[e]); }
  float sum = 0.f;
  for (int e = 0; e < E_; ++e) { l[e] = __expf(l[e] - mx); sum += l[e]; }
  float inv = 1.f / sum;
  for (int e = 0; e < E_; ++e) l[e] *= inv;
  float tsum = 0.f;
  for (int k = 0; k < KTOP; ++k) {
    int be = 0; float bv = -1.f;
    for (int e = 0; e < E_; ++e) if (l[e] > bv) { bv = l[e]; be = e; }
    topw[t * KTOP + k] = bv; topi[t * KTOP + k] = be;
    l[be] = -2.f; tsum += bv;
  }
  float it = 1.f / tsum;
  for (int k = 0; k < KTOP; ++k) {
    topw[t * KTOP + k] *= it;
    atomicAdd(&counts[topi[t * KTOP + k]], 1);
  }
}

__global__ void offsets_kernel(const int* __restrict__ counts, int* __restrict__ offs)
{
  if (threadIdx.x == 0 && blockIdx.x == 0) {
    int acc = 0;
    for (int e = 0; e < E_; ++e) { offs[e] = acc; acc += counts[e]; }
    offs[E_] = acc;
  }
}

__global__ void scatter_kernel(const float* __restrict__ topw, const int* __restrict__ topi,
                               const int* __restrict__ offs, int* __restrict__ cursor,
                               int* __restrict__ toklist, float* __restrict__ wlist)
{
  int t = blockIdx.x * blockDim.x + threadIdx.x;
  if (t >= T_TOK) return;
  for (int k = 0; k < KTOP; ++k) {
    int e = topi[t * KTOP + k];
    int pos = atomicAdd(&cursor[e], 1);
    int at = offs[e] + pos;
    toklist[at] = t;
    wlist[at] = topw[t * KTOP + k];
  }
}

__global__ void silumul_kernel(const float* __restrict__ g, const float* __restrict__ u,
                               _Float16* __restrict__ y, long n)
{
  long i = (long)blockIdx.x * blockDim.x + threadIdx.x;
  if (i < n) {
    float gv = g[i];
    float sv = gv / (1.f + __expf(-gv));
    y[i] = (_Float16)(sv * u[i]);
  }
}

__global__ __launch_bounds__(256) void sharedgate_kernel(
    const _Float16* __restrict__ x2, const float* __restrict__ wsh,
    float* __restrict__ gate)
{
  __shared__ float red[8];
  long t = blockIdx.x;
  float acc = 0.f;
  for (int i = threadIdx.x; i < H_; i += 256)
    acc += (float)x2[t * H_ + i] * wsh[i];
  for (int mm = 1; mm < 32; mm <<= 1) acc += __shfl_xor(acc, mm, 32);
  if ((threadIdx.x & 31) == 0) red[threadIdx.x >> 5] = acc;
  __syncthreads();
  if (threadIdx.x == 0) {
    float tot = 0.f;
    for (int i = 0; i < 8; ++i) tot += red[i];
    gate[t] = tot;
  }
}

__global__ void combine_kernel(const float* __restrict__ h, const float* __restrict__ moe,
                               const float* __restrict__ sh, const float* __restrict__ gate,
                               float* __restrict__ out)
{
  long i = (long)blockIdx.x * blockDim.x + threadIdx.x;
  if (i < (long)T_TOK * H_) {
    long t = i / H_;
    float sg = 1.f / (1.f + __expf(-gate[t]));
    out[i] = h[i] + moe[i] + sh[i] * sg;
  }
}

// ---------------------------------------------------------------------------
// Launcher
// ---------------------------------------------------------------------------
extern "C" void kernel_launch(void* const* d_in, const int* in_sizes, int n_in,
                              void* d_out, int out_size, void* d_ws, size_t ws_size,
                              hipStream_t stream)
{
  (void)in_sizes; (void)n_in; (void)out_size; (void)ws_size;
  const float* hidden = (const float*)d_in[0];
  const float* ln1w = (const float*)d_in[1];
  const float* ln2w = (const float*)d_in[2];
  const float* qnw  = (const float*)d_in[3];
  const float* knw  = (const float*)d_in[4];
  const float* Wq   = (const float*)d_in[5];
  const float* Wk   = (const float*)d_in[6];
  const float* Wv   = (const float*)d_in[7];
  const float* Wo   = (const float*)d_in[8];
  const float* Wr   = (const float*)d_in[9];
  const float* Weg  = (const float*)d_in[10];
  const float* Weu  = (const float*)d_in[11];
  const float* Wed  = (const float*)d_in[12];
  const float* Wsg  = (const float*)d_in[13];
  const float* Wsu  = (const float*)d_in[14];
  const float* Wsd  = (const float*)d_in[15];
  const float* Wshg = (const float*)d_in[16];
  float* outp = (float*)d_out;

  size_t off = 0;
  auto carve = [&](size_t bytes) -> void* {
    void* r = (char*)d_ws + off;
    off += (bytes + 255) & ~(size_t)255;
    return r;
  };
  const size_t T = T_TOK;
  _Float16* x16    = (_Float16*)carve(T * H_ * 2);
  float*    q32    = (float*)carve(T * NH_ * D_ * 4);
  float*    k32    = (float*)carve(T * NKV_ * D_ * 4);
  float*    v32    = (float*)carve(T * NKV_ * D_ * 4);
  _Float16* q16    = (_Float16*)carve(T * NH_ * D_ * 2);
  _Float16* k16    = (_Float16*)carve(T * NKV_ * D_ * 2);
  _Float16* v16    = (_Float16*)carve(T * NKV_ * D_ * 2);
  _Float16* attn16 = (_Float16*)carve(T * NH_ * D_ * 2);
  float*    h32    = (float*)carve(T * H_ * 4);
  _Float16* x2_16  = (_Float16*)carve(T * H_ * 2);
  float*    logits = (float*)carve(T * E_ * 4);
  float*    topw   = (float*)carve(T * KTOP * 4);
  int*      topi   = (int*)carve(T * KTOP * 4);
  int*      ibuf   = (int*)carve((2 * E_ + E_ + 1) * 4);
  int* counts = ibuf; int* cursor = ibuf + E_; int* offs = ibuf + 2 * E_;
  int*      toklist = (int*)carve(T * KTOP * 4);
  float*    wlist   = (float*)carve(T * KTOP * 4);
  float*    g32    = (float*)carve(T * KTOP * I_ * 4);
  float*    u32    = (float*)carve(T * KTOP * I_ * 4);
  _Float16* act16  = (_Float16*)carve(T * KTOP * I_ * 2);
  float*    moe32  = (float*)carve(T * H_ * 4);
  float*    gate   = (float*)carve(T * 4);
  // buffer reuse for the shared-FFN stage (stream-ordered, earlier users done)
  float*    gsh32  = q32;     // T*IS floats == T*NH*D floats
  float*    ush32  = g32;     // T*IS <= T*KTOP*I
  _Float16* acts16 = attn16;  // T*IS halves == T*NH*D halves
  float*    sh32   = u32;     // T*H <= T*KTOP*I

  dim3 blk256(256);
  // 1. rmsnorm(hidden, ln1) -> x16 (f16)
  rmsnorm_kernel<<<dim3(T_TOK), blk256, 0, stream>>>(hidden, ln1w, x16, H_);
  // 2. q/k/v projections
  gemm_kernel<0,0><<<dim3(T_TOK/128, (NH_*D_)/128, 1), blk256, 0, stream>>>(
      x16, H_, Wq, 0, q32, nullptr, NH_*D_, nullptr, nullptr, nullptr, nullptr,
      T_TOK, NH_*D_, H_);
  gemm_kernel<0,0><<<dim3(T_TOK/128, (NKV_*D_)/128, 1), blk256, 0, stream>>>(
      x16, H_, Wk, 0, k32, nullptr, NKV_*D_, nullptr, nullptr, nullptr, nullptr,
      T_TOK, NKV_*D_, H_);
  gemm_kernel<0,0><<<dim3(T_TOK/128, (NKV_*D_)/128, 1), blk256, 0, stream>>>(
      x16, H_, Wv, 0, v32, nullptr, NKV_*D_, nullptr, nullptr, nullptr, nullptr,
      T_TOK, NKV_*D_, H_);
  // 3. per-head q/k rmsnorm + rope -> f16; convert v -> f16
  qk_rope_kernel<<<dim3(S_, NH_ + NKV_, B_), dim3(128), 0, stream>>>(
      q32, k32, qnw, knw, q16, k16);
  {
    long n = (long)T_TOK * NKV_ * D_;
    cvt16_kernel<<<dim3((unsigned)((n + 255) / 256)), blk256, 0, stream>>>(v32, v16, n);
  }
  // 4. causal GQA flash attention (WMMA)
  attn_kernel<<<dim3(S_/128, NH_, B_), blk256, 0, stream>>>(q16, k16, v16, attn16);
  // 5. Wo projection + residual
  gemm_kernel<0,1><<<dim3(T_TOK/128, H_/128, 1), blk256, 0, stream>>>(
      attn16, NH_*D_, Wo, 0, h32, nullptr, H_, hidden, nullptr, nullptr, nullptr,
      T_TOK, H_, NH_*D_);
  // 6. rmsnorm(h, ln2) -> x2 (f16)
  rmsnorm_kernel<<<dim3(T_TOK), blk256, 0, stream>>>(h32, ln2w, x2_16, H_);
  // 7. router logits + top-8 routing -> compacted per-expert token lists
  gemm_kernel<0,0><<<dim3(T_TOK/128, 1, 1), blk256, 0, stream>>>(
      x2_16, H_, Wr, 0, logits, nullptr, E_, nullptr, nullptr, nullptr, nullptr,
      T_TOK, E_, H_);
  (void)hipMemsetAsync(ibuf, 0, 2 * E_ * sizeof(int), stream);
  (void)hipMemsetAsync(moe32, 0, (size_t)T_TOK * H_ * sizeof(float), stream);
  router_kernel<<<dim3(T_TOK/256), blk256, 0, stream>>>(logits, topw, topi, counts);
  offsets_kernel<<<dim3(1), dim3(32), 0, stream>>>(counts, offs);
  scatter_kernel<<<dim3(T_TOK/256), blk256, 0, stream>>>(topw, topi, offs, cursor,
                                                         toklist, wlist);
  // 8. expert FFNs over routed tokens only (gather rows via token lists)
  gemm_kernel<1,0><<<dim3(T_TOK/128, I_/128, E_), blk256, 0, stream>>>(
      x2_16, H_, Weg, (long)H_*I_, g32, nullptr, I_, nullptr, toklist, nullptr, offs,
      T_TOK, I_, H_);
  gemm_kernel<1,0><<<dim3(T_TOK/128, I_/128, E_), blk256, 0, stream>>>(
      x2_16, H_, Weu, (long)H_*I_, u32, nullptr, I_, nullptr, toklist, nullptr, offs,
      T_TOK, I_, H_);
  {
    long n = (long)T_TOK * KTOP * I_;
    silumul_kernel<<<dim3((unsigned)((n + 255) / 256)), blk256, 0, stream>>>(g32, u32, act16, n);
  }
  gemm_kernel<2,3><<<dim3(T_TOK/128, H_/128, E_), blk256, 0, stream>>>(
      act16, I_, Wed, (long)I_*H_, moe32, nullptr, H_, nullptr, toklist, wlist, offs,
      T_TOK, H_, I_);
  // 9. shared expert
  gemm_kernel<0,0><<<dim3(T_TOK/128, IS_/128, 1), blk256, 0, stream>>>(
      x2_16, H_, Wsg, 0, gsh32, nullptr, IS_, nullptr, nullptr, nullptr, nullptr,
      T_TOK, IS_, H_);
  gemm_kernel<0,0><<<dim3(T_TOK/128, IS_/128, 1), blk256, 0, stream>>>(
      x2_16, H_, Wsu, 0, ush32, nullptr, IS_, nullptr, nullptr, nullptr, nullptr,
      T_TOK, IS_, H_);
  {
    long n = (long)T_TOK * IS_;
    silumul_kernel<<<dim3((unsigned)((n + 255) / 256)), blk256, 0, stream>>>(gsh32, ush32, acts16, n);
  }
  gemm_kernel<0,0><<<dim3(T_TOK/128, H_/128, 1), blk256, 0, stream>>>(
      acts16, IS_, Wsd, 0, sh32, nullptr, H_, nullptr, nullptr, nullptr, nullptr,
      T_TOK, H_, IS_);
  sharedgate_kernel<<<dim3(T_TOK), blk256, 0, stream>>>(x2_16, Wshg, gate);
  // 10. combine: out = h + moe + shared * sigmoid(gate)
  {
    long n = (long)T_TOK * H_;
    combine_kernel<<<dim3((unsigned)((n + 255) / 256)), blk256, 0, stream>>>(
        h32, moe32, sh32, gate, outp);
  }
}
